// TwoStageNet_75917841924734
// MI455X (gfx1250) — matrix operand order repaired and checked by reading, not compiled, not run on hardware
//
#include <hip/hip_runtime.h>

// ---------------- problem constants ----------------
#define B_    4
#define C_    128
#define H_    128
#define W_    128
#define HW_   (H_ * W_)
#define K_    1500
#define N_    (B_ * K_)
#define NPAD_ 6016                // N_ padded to multiple of 128 for async GEMM tiles
#define HID_  1024
#define FCK_  (C_ * 9)            // 1152
#define NMS_THR_ 0.7f

// ---------------- d_out flat offsets (elements) ----------------
#define OFF_HM0  0
#define OFF_HM1  65536
#define OFF_WH0  131072
#define OFF_WH1  262144
#define OFF_OF0  393216
#define OFF_OF1  524288
#define OFF_CLS  655360
#define OFF_REG  679360
#define OFF_BXY  703360
#define OFF_SCR  733360
#define OFF_KEEP 739360

// ---------------- workspace byte offsets ----------------
#define WS_FEATP 0u               // BHWC feat1, 8,388,608 f32
#define WS_BT    33554432u        // w_fc^T bf16 [1024][1152]
#define WS_A     35913728u        // ROI A bf16 [6016][1152] (16 pad rows zeroed)
#define WS_H     49774592u        // h f32 [6000][1024]
#define WS_IDX   74350592u        // topk idx int [6000]
#define WS_VAL   74374592u        // topk val f32 [6000]
#define WS_BOX   74398592u        // boxes f32 [6000][4]
#define WS_KEEP  74494592u        // keep int [6000]

typedef __attribute__((ext_vector_type(16))) __bf16        v16bf;
typedef __attribute__((ext_vector_type(8)))  float         v8f;
typedef __attribute__((ext_vector_type(8)))  unsigned int  v8u;

// ---------------- helpers ----------------
__device__ __forceinline__ unsigned short f2bf(float f) {
  unsigned u = __builtin_bit_cast(unsigned, f);
  unsigned r = u + 0x7FFFu + ((u >> 16) & 1u);   // round-to-nearest-even
  return (unsigned short)(r >> 16);
}
__device__ __forceinline__ unsigned fkey(float f) {   // order-preserving float->uint
  unsigned u = __builtin_bit_cast(unsigned, f);
  return (u & 0x80000000u) ? ~u : (u | 0x80000000u);
}
__device__ __forceinline__ float unfkey(unsigned k) {
  unsigned u = (k & 0x80000000u) ? (k ^ 0x80000000u) : ~k;
  return __builtin_bit_cast(float, u);
}
__device__ __forceinline__ unsigned lds_off32(const void* p) {
  // AS(3)->generic addrspacecast keeps the LDS byte offset in the low 32 bits
  return (unsigned)(unsigned long long)p;
}
__device__ __forceinline__ void async_load_b128(unsigned lds_byte_off,
                                                unsigned gl_byte_off,
                                                unsigned long long base) {
  asm volatile("global_load_async_to_lds_b128 %0, %1, %2"
               :: "v"(lds_byte_off), "v"(gl_byte_off), "s"(base)
               : "memory");
}
__device__ __forceinline__ void wait_asynccnt0() {
  asm volatile("s_wait_asynccnt 0x0" ::: "memory");
}

// ---------------- 1a: convert w_fc -> bf16, transposed [N][K] ----------------
__global__ void wfc_convert_kernel(const float* __restrict__ w_fc,
                                   unsigned short* __restrict__ Bt) {
  int i = blockIdx.x * 256 + threadIdx.x;           // n-major over 1024*1152
  if (i >= HID_ * FCK_) return;
  int n = i / FCK_, k = i - n * FCK_;
  Bt[i] = f2bf(w_fc[(size_t)k * HID_ + n]);
}

// ---------------- 1b: zero the 16 pad rows of A ----------------
__global__ void zeropad_kernel(unsigned short* __restrict__ Abf) {
  int i = blockIdx.x * 256 + threadIdx.x;           // 16*1152 = 18432
  if (i < (NPAD_ - N_) * FCK_) Abf[(size_t)N_ * FCK_ + i] = 0;
}

// ---------------- 2: conv1x1 heads (hm/wh/off for both feats) ----------------
__global__ __launch_bounds__(256) void heads_kernel(
    const float* __restrict__ feat0, const float* __restrict__ feat1,
    const float* __restrict__ w_hm0, const float* __restrict__ b_hm0,
    const float* __restrict__ w_hm1, const float* __restrict__ b_hm1,
    const float* __restrict__ w_wh0, const float* __restrict__ b_wh0,
    const float* __restrict__ w_wh1, const float* __restrict__ b_wh1,
    const float* __restrict__ w_of0, const float* __restrict__ b_of0,
    const float* __restrict__ w_of1, const float* __restrict__ b_of1,
    float* __restrict__ out) {
  __shared__ float s[1280];
  int tid = threadIdx.x;
  if (tid < 128) { s[tid] = w_hm0[tid]; s[128 + tid] = w_hm1[tid]; }
  s[256 + tid]  = w_wh0[tid];
  s[512 + tid]  = w_wh1[tid];
  s[768 + tid]  = w_of0[tid];
  s[1024 + tid] = w_of1[tid];
  __syncthreads();
  int p = blockIdx.x * 256 + tid;
  int b = p >> 14, pix = p & (HW_ - 1);
  float a0 = 0, a1 = 0, wx0 = 0, wy0 = 0, wx1 = 0, wy1 = 0,
        ox0 = 0, oy0 = 0, ox1 = 0, oy1 = 0;
  const float* f0p = feat0 + (size_t)b * C_ * HW_ + pix;
  const float* f1p = feat1 + (size_t)b * C_ * HW_ + pix;
  for (int c = 0; c < C_; ++c) {
    float f0 = f0p[(size_t)c * HW_];
    float f1 = f1p[(size_t)c * HW_];
    a0  += f0 * s[c];            a1  += f1 * s[128 + c];
    wx0 += f0 * s[256 + c];      wy0 += f0 * s[384 + c];
    wx1 += f1 * s[512 + c];      wy1 += f1 * s[640 + c];
    ox0 += f0 * s[768 + c];      oy0 += f0 * s[896 + c];
    ox1 += f1 * s[1024 + c];     oy1 += f1 * s[1152 + c];
  }
  out[OFF_HM0 + p] = a0 + b_hm0[0];
  out[OFF_HM1 + p] = a1 + b_hm1[0];
  size_t wb = (size_t)b * 2 * HW_ + pix;
  out[OFF_WH0 + wb]       = wx0 + b_wh0[0];
  out[OFF_WH0 + wb + HW_] = wy0 + b_wh0[1];
  out[OFF_WH1 + wb]       = wx1 + b_wh1[0];
  out[OFF_WH1 + wb + HW_] = wy1 + b_wh1[1];
  out[OFF_OF0 + wb]       = ox0 + b_of0[0];
  out[OFF_OF0 + wb + HW_] = oy0 + b_of0[1];
  out[OFF_OF1 + wb]       = ox1 + b_of1[0];
  out[OFF_OF1 + wb + HW_] = oy1 + b_of1[1];
}

// ---------------- 3: feat1 BCHW -> BHWC transpose (LDS tiled) ----------------
__global__ __launch_bounds__(256) void transpose_kernel(
    const float* __restrict__ in, float* __restrict__ out) {
  __shared__ float t[32][33];
  int xt = blockIdx.x * 32, ct = blockIdx.y * 32;
  int by = blockIdx.z;
  int b = by >> 7, y = by & (H_ - 1);
  int tx = threadIdx.x & 31, ty = threadIdx.x >> 5;
  const float* ip = in + (size_t)b * C_ * HW_ + (size_t)y * W_;
  for (int r = ty; r < 32; r += 8)
    t[r][tx] = ip[(size_t)(ct + r) * HW_ + xt + tx];
  __syncthreads();
  float* op = out + ((size_t)b * H_ + y) * W_ * C_;
  for (int r = ty; r < 32; r += 8)
    op[(size_t)(xt + r) * C_ + ct + tx] = t[tx][r];
}

// ---------------- 4: exact top-K (radix threshold + bitonic sort) ----------------
__global__ __launch_bounds__(1024) void topk_kernel(
    const float* __restrict__ hm1, float* __restrict__ out_val,
    int* __restrict__ out_idx) {
  const int b = blockIdx.x, tid = threadIdx.x;
  const float* src = hm1 + (size_t)b * HW_;
  __shared__ unsigned scnt;
  __shared__ unsigned ncand;
  __shared__ unsigned long long cand[2048];

  // K-th largest key via bitwise binary search
  unsigned prefix = 0u;
  for (int bit = 31; bit >= 0; --bit) {
    unsigned cv = prefix | (1u << bit);
    if (tid == 0) scnt = 0;
    __syncthreads();
    unsigned local = 0;
    for (int i = tid; i < HW_; i += 1024)
      local += (fkey(src[i]) >= cv) ? 1u : 0u;
    for (int s = 16; s > 0; s >>= 1) local += __shfl_xor(local, s, 32);
    if ((tid & 31) == 0) atomicAdd(&scnt, local);
    __syncthreads();
    if (scnt >= (unsigned)K_) prefix = cv;
    __syncthreads();
  }
  // gather candidates (key >= kth-largest)
  for (int i = tid; i < 2048; i += 1024) cand[i] = 0ull;
  if (tid == 0) ncand = 0;
  __syncthreads();
  for (int i = tid; i < HW_; i += 1024) {
    unsigned kk = fkey(src[i]);
    if (kk >= prefix) {
      unsigned pos = atomicAdd(&ncand, 1u);
      if (pos < 2048u)
        cand[pos] = ((unsigned long long)kk << 32) |
                    (unsigned long long)(0xFFFFFFFFu - (unsigned)i);
    }
  }
  __syncthreads();
  // bitonic sort 2048, descending (key desc, idx asc via ~idx)
  for (int k = 2; k <= 2048; k <<= 1) {
    for (int j = k >> 1; j > 0; j >>= 1) {
      for (int i = tid; i < 2048; i += 1024) {
        int ixj = i ^ j;
        if (ixj > i) {
          unsigned long long a = cand[i], c = cand[ixj];
          bool desc = ((i & k) == 0);
          if (desc ? (a < c) : (a > c)) { cand[i] = c; cand[ixj] = a; }
        }
      }
      __syncthreads();
    }
  }
  for (int i = tid; i < K_; i += 1024) {
    unsigned long long e = cand[i];
    out_val[b * K_ + i] = unfkey((unsigned)(e >> 32));
    out_idx[b * K_ + i] = (int)(0xFFFFFFFFu - (unsigned)(e & 0xFFFFFFFFu));
  }
}

// ---------------- 5: gather wh/off, build boxes ----------------
__global__ void boxes_kernel(const float* __restrict__ out,
                             const int* __restrict__ idx,
                             float* __restrict__ boxes) {
  int n = blockIdx.x * 256 + threadIdx.x;
  if (n >= N_) return;
  int b = n / K_;
  int id = idx[n];
  float xs = (float)(id & (W_ - 1));
  float ys = (float)(id >> 7);
  size_t base = (size_t)b * 2 * HW_ + id;
  float w0 = out[OFF_WH1 + base], w1 = out[OFF_WH1 + base + HW_];
  float o0 = out[OFF_OF1 + base], o1 = out[OFF_OF1 + base + HW_];
  float wc0 = fmaxf(w0, 0.f), wc1 = fmaxf(w1, 0.f);
  float px = xs + o0, py = ys + o1;
  float x1 = px - wc0 * 0.5f, y1 = py - wc1 * 0.5f;
  boxes[n * 4 + 0] = x1;
  boxes[n * 4 + 1] = y1;
  boxes[n * 4 + 2] = x1 + wc0;
  boxes[n * 4 + 3] = y1 + wc1;
}

// ---------------- 6: sequential NMS, one workgroup per batch ----------------
__global__ __launch_bounds__(1024) void nms_kernel(
    const float* __restrict__ boxes, int* __restrict__ keep) {
  __shared__ float sx1[K_], sy1[K_], sx2[K_], sy2[K_], sar[K_];
  __shared__ int kp[K_];
  int b = blockIdx.x, tid = threadIdx.x;
  const float* bp = boxes + (size_t)b * K_ * 4;
  for (int i = tid; i < K_; i += 1024) {
    float x1 = bp[i * 4 + 0], y1 = bp[i * 4 + 1];
    float x2 = bp[i * 4 + 2], y2 = bp[i * 4 + 3];
    sx1[i] = x1; sy1[i] = y1; sx2[i] = x2; sy2[i] = y2;
    sar[i] = fmaxf(x2 - x1, 0.f) * fmaxf(y2 - y1, 0.f);
    kp[i] = 1;
  }
  for (int i = 0; i < K_; ++i) {
    __syncthreads();
    if (!kp[i]) continue;                      // uniform
    float ax1 = sx1[i], ay1 = sy1[i], ax2 = sx2[i], ay2 = sy2[i], aar = sar[i];
    for (int j = i + 1 + tid; j < K_; j += 1024) {
      if (kp[j]) {
        float iw = fmaxf(fminf(ax2, sx2[j]) - fmaxf(ax1, sx1[j]), 0.f);
        float ih = fmaxf(fminf(ay2, sy2[j]) - fmaxf(ay1, sy1[j]), 0.f);
        float inter = iw * ih;
        float iou = inter / (aar + sar[j] - inter + 1e-9f);
        if (iou > NMS_THR_) kp[j] = 0;
      }
    }
  }
  __syncthreads();
  for (int i = tid; i < K_; i += 1024) keep[b * K_ + i] = kp[i];
}

// ---------------- 7: ROI align -> bf16 A matrix rows ----------------
__global__ __launch_bounds__(128) void roi_kernel(
    const float* __restrict__ featp, const float* __restrict__ boxes,
    unsigned short* __restrict__ Abf) {
  int n = blockIdx.x;
  int c = threadIdx.x;                 // channel
  int b = n / K_;
  float x1 = boxes[n * 4 + 0], y1 = boxes[n * 4 + 1];
  float x2 = boxes[n * 4 + 2], y2 = boxes[n * 4 + 3];
  float bw = fmaxf(x2 - x1, 1.f) * (1.f / 3.f);
  float bh = fmaxf(y2 - y1, 1.f) * (1.f / 3.f);
  float acc[9];
#pragma unroll
  for (int i = 0; i < 9; ++i) acc[i] = 0.f;
  const float* fp = featp + (size_t)b * HW_ * C_ + c;
  for (int sy = 0; sy < 6; ++sy) {
    float offy = (float)(sy >> 1) + ((float)(sy & 1) + 0.5f) * 0.5f;
    float yy = y1 + offy * bh;
    for (int sx = 0; sx < 6; ++sx) {
      float offx = (float)(sx >> 1) + ((float)(sx & 1) + 0.5f) * 0.5f;
      float xx = x1 + offx * bw;
      bool valid = (yy > -1.f) && (yy < (float)H_) && (xx > -1.f) && (xx < (float)W_);
      float y = fminf(fmaxf(yy, 0.f), (float)(H_ - 1));
      float x = fminf(fmaxf(xx, 0.f), (float)(W_ - 1));
      float yf = floorf(y), xf = floorf(x);
      int y0 = (int)yf, x0 = (int)xf;
      int y1i = min(y0 + 1, H_ - 1), x1i = min(x0 + 1, W_ - 1);
      float ly = y - yf, lx = x - xf;
      float hy = 1.f - ly, hx = 1.f - lx;
      float v00 = fp[((size_t)y0 * W_ + x0) * C_];
      float v01 = fp[((size_t)y0 * W_ + x1i) * C_];
      float v10 = fp[((size_t)y1i * W_ + x0) * C_];
      float v11 = fp[((size_t)y1i * W_ + x1i) * C_];
      float v = hy * hx * v00 + hy * lx * v01 + ly * hx * v10 + ly * lx * v11;
      acc[(sy >> 1) * 3 + (sx >> 1)] += valid ? v : 0.f;
    }
  }
  unsigned short* arow = Abf + (size_t)n * FCK_ + c * 9;
#pragma unroll
  for (int i = 0; i < 9; ++i) arow[i] = f2bf(acc[i] * 0.25f);
}

// ---------------- 8: FC GEMM h = relu(A @ w_fc + b) via bf16 WMMA ----------------
// A: [6016][1152] bf16 row-major.  Bt: [1024][1152] bf16 (N-major).
// Block tile 128x128, 8 waves; wave = 4(M) x 2(N) WMMA 16x16 tiles, K-step 32.
// Double-buffered LDS staged by GLOBAL_LOAD_ASYNC_TO_LDS_B128 (ASYNCcnt).
__global__ __launch_bounds__(256) void fc_gemm_wmma(
    const unsigned short* __restrict__ A, const unsigned short* __restrict__ Bt,
    const float* __restrict__ bias, float* __restrict__ Hout) {
  __shared__ __align__(16) unsigned int lA[2][2048];  // 128 rows x 32 bf16 per buf
  __shared__ __align__(16) unsigned int lB[2][2048];  // 128 cols x 32 bf16 per buf
  const int tid = threadIdx.x;
  const int lane = tid & 31, wid = tid >> 5;
  const int waveM = wid & 1, waveN = wid >> 1;
  const int half = lane >> 4, l15 = lane & 15;
  const int rowBase = blockIdx.x * 128;
  const int colBase = blockIdx.y * 128;

  const unsigned ldsA0 = lds_off32(&lA[0][0]);
  const unsigned ldsB0 = lds_off32(&lB[0][0]);
  const unsigned long long Abase = (unsigned long long)A;
  const unsigned long long Bbase = (unsigned long long)Bt;

  // issue one 128x32 bf16 tile pair into LDS buffer `buf` via async DMA
  auto issueTile = [&](int k0, int buf) {
#pragma unroll
    for (int q = tid; q < 512; q += 256) {           // 512 x 16B chunks per tile
      int r = q >> 2, c4 = q & 3;                    // row, 16B-chunk within row
      unsigned lrow = (unsigned)(r * 64 + c4 * 16);  // LDS bytes (row = 64B)
      unsigned goffA = ((unsigned)(rowBase + r) * FCK_ + (unsigned)k0 + c4 * 8) * 2u;
      async_load_b128(ldsA0 + (unsigned)buf * 8192u + lrow, goffA, Abase);
      unsigned goffB = ((unsigned)(colBase + r) * FCK_ + (unsigned)k0 + c4 * 8) * 2u;
      async_load_b128(ldsB0 + (unsigned)buf * 8192u + lrow, goffB, Bbase);
    }
  };

  v8f acc[4][2];
#pragma unroll
  for (int mm = 0; mm < 4; ++mm)
#pragma unroll
    for (int nn = 0; nn < 2; ++nn) acc[mm][nn] = (v8f)0.0f;

  const int NT = FCK_ / 32;                          // 36 K-steps
  issueTile(0, 0);
  for (int t = 0; t < NT; ++t) {
    wait_asynccnt0();        // my DMA writes into buf (t&1) are complete
    __syncthreads();         // everyone's writes done; prev reads of other buf done
    if (t + 1 < NT) issueTile((t + 1) * 32, (t + 1) & 1);
    const unsigned int* bufA = lA[t & 1];
    const unsigned int* bufB = lB[t & 1];

    v16bf bfrag[2];
#pragma unroll
    for (int nn = 0; nn < 2; ++nn) {
      int nloc = waveN * 32 + nn * 16 + l15;
      v8u bu;
#pragma unroll
      for (int p = 0; p < 8; ++p) bu[p] = bufB[nloc * 16 + half * 8 + p];
      bfrag[nn] = __builtin_bit_cast(v16bf, bu);
    }
#pragma unroll
    for (int mm = 0; mm < 4; ++mm) {
      int mloc = waveM * 64 + mm * 16 + l15;
      v8u au;
#pragma unroll
      for (int p = 0; p < 4; ++p) {
        au[p]     = bufA[mloc * 16 + half * 8 + p];      // K = half*16 + 2p
        au[p + 4] = bufA[mloc * 16 + half * 8 + 4 + p];  // K = half*16 + 8 + 2p
      }
      v16bf afrag = __builtin_bit_cast(v16bf, au);
#pragma unroll
      for (int nn = 0; nn < 2; ++nn) {
        acc[mm][nn] = __builtin_amdgcn_wmma_f32_16x16x32_bf16(
            false, afrag, false, bfrag[nn], (short)0, acc[mm][nn], false, false);
      }
    }
  }
  // epilogue: bias + relu, C/D layout lane: N=l&15, VGPR r: M=r+8*half
#pragma unroll
  for (int mm = 0; mm < 4; ++mm) {
#pragma unroll
    for (int nn = 0; nn < 2; ++nn) {
      int col = colBase + waveN * 32 + nn * 16 + l15;
      float bv = bias[col];
#pragma unroll
      for (int r = 0; r < 8; ++r) {
        int row = rowBase + waveM * 64 + mm * 16 + r + half * 8;
        float v = acc[mm][nn][r] + bv;
        v = v > 0.f ? v : 0.f;
        if (row < N_) Hout[(size_t)row * HID_ + col] = v;
      }
    }
  }
}

// ---------------- 9: cls/reg heads (wave-per-row dot products) ----------------
__global__ __launch_bounds__(256) void clsreg_kernel(
    const float* __restrict__ Hm, const float* __restrict__ w_cls,
    const float* __restrict__ b_cls, const float* __restrict__ w_reg,
    const float* __restrict__ b_reg, const int* __restrict__ keep,
    float* __restrict__ out) {
  int wid = threadIdx.x >> 5, lane = threadIdx.x & 31;
  int row = blockIdx.x * 8 + wid;
  if (row >= N_) return;
  float acc[8] = {0, 0, 0, 0, 0, 0, 0, 0};
  const float* hp = Hm + (size_t)row * HID_;
  for (int k = lane; k < HID_; k += 32) {
    float hv = hp[k];
    float4 wc = ((const float4*)w_cls)[k];
    float4 wr = ((const float4*)w_reg)[k];
    acc[0] += hv * wc.x; acc[1] += hv * wc.y; acc[2] += hv * wc.z; acc[3] += hv * wc.w;
    acc[4] += hv * wr.x; acc[5] += hv * wr.y; acc[6] += hv * wr.z; acc[7] += hv * wr.w;
  }
#pragma unroll
  for (int i = 0; i < 8; ++i)
    for (int s = 16; s > 0; s >>= 1) acc[i] += __shfl_xor(acc[i], s, 32);
  if (lane == 0) {
    float m = keep[row] ? 1.f : 0.f;
    for (int i = 0; i < 4; ++i) {
      out[OFF_CLS + row * 4 + i] = (acc[i] + b_cls[i]) * m;
      out[OFF_REG + row * 4 + i] = (acc[4 + i] + b_reg[i]) * m;
    }
  }
}

// ---------------- 10: bxyxys, scores, keep ----------------
__global__ void finalize_kernel(const float* __restrict__ boxes,
                                const float* __restrict__ val,
                                const int* __restrict__ keep,
                                float* __restrict__ out) {
  int n = blockIdx.x * 256 + threadIdx.x;
  if (n >= N_) return;
  float m = keep[n] ? 1.f : 0.f;
  out[OFF_BXY + n * 5 + 0] = (float)(n / K_) * m;
  out[OFF_BXY + n * 5 + 1] = boxes[n * 4 + 0] * m;
  out[OFF_BXY + n * 5 + 2] = boxes[n * 4 + 1] * m;
  out[OFF_BXY + n * 5 + 3] = boxes[n * 4 + 2] * m;
  out[OFF_BXY + n * 5 + 4] = boxes[n * 4 + 3] * m;
  out[OFF_SCR + n] = val[n] * m;
  out[OFF_KEEP + n] = m;
}

// ---------------- launch ----------------
extern "C" void kernel_launch(void* const* d_in, const int* in_sizes, int n_in,
                              void* d_out, int out_size, void* d_ws, size_t ws_size,
                              hipStream_t stream) {
  const float* feat0 = (const float*)d_in[0];
  const float* feat1 = (const float*)d_in[1];
  const float* w_hm0 = (const float*)d_in[2];
  const float* b_hm0 = (const float*)d_in[3];
  const float* w_hm1 = (const float*)d_in[4];
  const float* b_hm1 = (const float*)d_in[5];
  const float* w_wh0 = (const float*)d_in[6];
  const float* b_wh0 = (const float*)d_in[7];
  const float* w_wh1 = (const float*)d_in[8];
  const float* b_wh1 = (const float*)d_in[9];
  const float* w_of0 = (const float*)d_in[10];
  const float* b_of0 = (const float*)d_in[11];
  const float* w_of1 = (const float*)d_in[12];
  const float* b_of1 = (const float*)d_in[13];
  const float* w_fc  = (const float*)d_in[14];
  const float* b_fc  = (const float*)d_in[15];
  const float* w_cls = (const float*)d_in[16];
  const float* b_cls = (const float*)d_in[17];
  const float* w_reg = (const float*)d_in[18];
  const float* b_reg = (const float*)d_in[19];

  char* ws = (char*)d_ws;
  float*          featp = (float*)(ws + WS_FEATP);
  unsigned short* Bt    = (unsigned short*)(ws + WS_BT);
  unsigned short* Abf   = (unsigned short*)(ws + WS_A);
  float*          Hm    = (float*)(ws + WS_H);
  int*            tidx  = (int*)(ws + WS_IDX);
  float*          tval  = (float*)(ws + WS_VAL);
  float*          boxes = (float*)(ws + WS_BOX);
  int*            keepi = (int*)(ws + WS_KEEP);
  float*          out   = (float*)d_out;

  wfc_convert_kernel<<<(HID_ * FCK_ + 255) / 256, 256, 0, stream>>>(w_fc, Bt);

  zeropad_kernel<<<((NPAD_ - N_) * FCK_ + 255) / 256, 256, 0, stream>>>(Abf);

  heads_kernel<<<(B_ * HW_) / 256, 256, 0, stream>>>(
      feat0, feat1, w_hm0, b_hm0, w_hm1, b_hm1, w_wh0, b_wh0, w_wh1, b_wh1,
      w_of0, b_of0, w_of1, b_of1, out);

  transpose_kernel<<<dim3(W_ / 32, C_ / 32, B_ * H_), 256, 0, stream>>>(feat1, featp);

  topk_kernel<<<B_, 1024, 0, stream>>>(out + OFF_HM1, tval, tidx);

  boxes_kernel<<<(N_ + 255) / 256, 256, 0, stream>>>(out, tidx, boxes);

  nms_kernel<<<B_, 1024, 0, stream>>>(boxes, keepi);

  roi_kernel<<<N_, 128, 0, stream>>>(featp, boxes, Abf);

  fc_gemm_wmma<<<dim3(NPAD_ / 128, HID_ / 128), 256, 0, stream>>>(
      Abf, Bt, b_fc, Hm);

  clsreg_kernel<<<N_ / 8, 256, 0, stream>>>(Hm, w_cls, b_cls, w_reg, b_reg, keepi, out);

  finalize_kernel<<<(N_ + 255) / 256, 256, 0, stream>>>(boxes, tval, keepi, out);
}
